// MultiHeadAttention_6992206758379
// MI455X (gfx1250) — matrix operand order repaired and checked by reading, not compiled, not run on hardware
//
#include <hip/hip_runtime.h>
#include <hip/hip_bf16.h>

typedef __attribute__((ext_vector_type(4)))  __bf16 v4bf;
typedef __attribute__((ext_vector_type(8)))  __bf16 v8bf;
typedef __attribute__((ext_vector_type(16))) __bf16 v16bf;
typedef __attribute__((ext_vector_type(8)))  float  v8f;

#define DIM_   2048
#define SEQ_   2048
#define NH_    16
#define HD_    128
#define WSEL_  8388608   // elements per workspace tensor (B*H*S*hd = B*S*DIM)

#if defined(__has_builtin)
#if __has_builtin(__builtin_amdgcn_global_load_async_to_lds_b128)
#define HAVE_ASYNC_LDS 1
#endif
#endif
#ifndef HAVE_ASYNC_LDS
#define HAVE_ASYNC_LDS 0
#endif

#if HAVE_ASYNC_LDS
typedef int vsi4 __attribute__((vector_size(16)));
typedef __attribute__((address_space(1))) vsi4 g_vsi4;
typedef __attribute__((address_space(3))) vsi4 l_vsi4;
#endif

__device__ __forceinline__ v16bf cat8(v8bf lo, v8bf hi) {
  v16bf r;
#pragma unroll
  for (int i = 0; i < 8; ++i) { r[i] = lo[i]; r[i + 8] = hi[i]; }
  return r;
}

__device__ __forceinline__ v4bf cvt4(float4 v) {
  v4bf r;
  r[0] = (__bf16)v.x; r[1] = (__bf16)v.y; r[2] = (__bf16)v.z; r[3] = (__bf16)v.w;
  return r;
}

// A fragment (16x32 bf16, MxK): lane&15 = M row; lanes<16 hold K {0..7,16..23},
// lanes>=16 hold K {8..15,24..31}.  (ISA 7.12.2, 16-bit A layout)
__device__ __forceinline__ v16bf load_a_frag(const __bf16* p0, int ld, int lane) {
  const int m  = lane & 15;
  const int kb = (lane < 16) ? 0 : 8;
  const __bf16* p = p0 + m * ld + kb;
  v8bf lo = *(const v8bf*)p;
  v8bf hi = *(const v8bf*)(p + 16);
  return cat8(lo, hi);
}

// B fragment (32x16 bf16, KxN), tile stored [n][k]: lane&15 = N col;
// lanes<16 hold K {0..7,16..23}, lanes>=16 hold K {8..15,24..31}.
__device__ __forceinline__ v16bf load_b_frag(const __bf16* p0, int ld, int lane) {
  const int n  = lane & 15;
  const int kb = (lane < 16) ? 0 : 8;
  const __bf16* p = p0 + n * ld + kb;
  v8bf lo = *(const v8bf*)p;
  v8bf hi = *(const v8bf*)(p + 16);
  return cat8(lo, hi);
}

__device__ __forceinline__ v8f wmma_bf16(v16bf a, v16bf b, v8f c) {
  return __builtin_amdgcn_wmma_f32_16x16x32_bf16(false, a, false, b, (short)0, c,
                                                 false, false);
}

// ---------------------------------------------------------------------------
// Kernel 1: Q/K/V projection GEMM (C = X * W^T) + fused RoPE epilogue.
// 128x128 block tile (N tile == one head), 8 waves, double-buffered LDS,
// register-staged global loads issued one K-step ahead, B-frags batch-loaded
// into distinct registers so ds latency hides behind the WMMA chain.
// ---------------------------------------------------------------------------
__global__ __launch_bounds__(256) void qkv_rope_kernel(
    const float* __restrict__ x,  const float* __restrict__ Wq,
    const float* __restrict__ Wk, const float* __restrict__ Wv,
    __bf16* __restrict__ Qo, __bf16* __restrict__ Ko, __bf16* __restrict__ Vo)
{
  constexpr int LD = 40;                 // 32 + 8 pad (80B rows, 16B aligned)
  __shared__ __bf16 sA[2][128][LD];
  __shared__ __bf16 sB[2][128][LD];

  const int z  = blockIdx.z;
  const float* W = (z == 0) ? Wq : (z == 1) ? Wk : Wv;
  __bf16* Out    = (z == 0) ? Qo : (z == 1) ? Ko : Vo;

  const int m0   = blockIdx.x * 128;     // row base in (B*S)
  const int h    = blockIdx.y;           // head == N tile
  const int n0   = h * HD_;
  const int tid  = threadIdx.x;
  const int lane = tid & 31;
  const int wave = tid >> 5;

  v8f acc[8];
#pragma unroll
  for (int i = 0; i < 8; ++i)
#pragma unroll
    for (int j = 0; j < 8; ++j) acc[i][j] = 0.f;

  const int rs = tid >> 3;               // 0..31 (staging row group)
  const int c4 = (tid & 7) * 4;          // 0..28 (staging col)
  const float* xrow = x + (size_t)(m0 + rs) * DIM_ + c4;
  const float* wrow = W + (size_t)(n0 + rs) * DIM_ + c4;

  float4 ra[4], rb[4];
  auto issue = [&](int k0) {
#pragma unroll
    for (int it = 0; it < 4; ++it) {
      ra[it] = *(const float4*)(xrow + (size_t)it * 32 * DIM_ + k0);
      rb[it] = *(const float4*)(wrow + (size_t)it * 32 * DIM_ + k0);
    }
  };

  issue(0);
  int p = 0;
  for (int kt = 0; kt < DIM_ / 32; ++kt) {
#pragma unroll
    for (int it = 0; it < 4; ++it) {
      const int row = rs + it * 32;
      *(v4bf*)&sA[p][row][c4] = cvt4(ra[it]);
      *(v4bf*)&sB[p][row][c4] = cvt4(rb[it]);
    }
    if (kt + 1 < DIM_ / 32) issue((kt + 1) * 32);   // prefetch next tile
    __syncthreads();
    const v16bf a = load_a_frag(&sA[p][wave * 16][0], LD, lane);
    v16bf bfr[8];
#pragma unroll
    for (int nt = 0; nt < 8; ++nt)
      bfr[nt] = load_b_frag(&sB[p][nt * 16][0], LD, lane);
#pragma unroll
    for (int nt = 0; nt < 8; ++nt)
      acc[nt] = wmma_bf16(a, bfr[nt], acc[nt]);
    p ^= 1;
  }

  // Epilogue: RoPE (Q,K) then store bf16 into (B,H,S,hd).
  const int lh = (lane < 16) ? 0 : 8;
  const int nc = lane & 15;
#pragma unroll
  for (int r = 0; r < 8; ++r) {
    const int m = m0 + wave * 16 + r + lh;      // global row in B*S
    const int b = m >> 11;
    const int s = m & (SEQ_ - 1);
#pragma unroll
    for (int nt = 0; nt < 8; ++nt) {
      const int d = nt * 16 + nc;               // head-dim index 0..127
      const float v = acc[nt][r];
      float o;
      if (z < 2) {
        // inv_freq = 10000^{-(d%64)/64}; partner d^64 is same lane, tile nt^4.
        const float inv = __expf(-(float)(d & 63) * 0.14391156831212787f);
        float sn, cs;
        __sincosf((float)s * inv, &sn, &cs);
        const float partner = acc[nt ^ 4][r];
        o = (d < 64) ? (v * cs - partner * sn) : (v * cs + partner * sn);
      } else {
        o = v;
      }
      Out[(((size_t)b * NH_ + h) * SEQ_ + s) * HD_ + d] = (__bf16)o;
    }
  }
}

// ---------------------------------------------------------------------------
// Kernel 2: causal flash attention, one (b,h,128-q-row tile) per workgroup.
// ---------------------------------------------------------------------------
__global__ __launch_bounds__(256) void flash_attn_kernel(
    const __bf16* __restrict__ Q, const __bf16* __restrict__ K,
    const __bf16* __restrict__ V, __bf16* __restrict__ O)
{
  constexpr int LDK = HD_ + 8;   // 136
  constexpr int LDV = 64 + 8;    // 72
  __shared__ __bf16 sK[64][LDK];        // K tile  [kv][d]
  __shared__ __bf16 sVt[HD_][LDV];      // V tile transposed [d][kv]
  __shared__ __bf16 sP[8][16][LDV];     // per-wave P (C/D -> A layout bridge)

  const int q0   = blockIdx.x * 128;
  const int h    = blockIdx.y;
  const int b    = blockIdx.z;
  const int tid  = threadIdx.x;
  const int lane = tid & 31;
  const int wave = tid >> 5;

  const size_t head_off = ((size_t)b * NH_ + h) * SEQ_;

  // Q fragments live in registers: 16 rows/wave, K=128 -> 4 chunks of 32.
  v16bf aq[4];
  {
    const __bf16* qrow = Q + (head_off + q0 + wave * 16 + (lane & 15)) * HD_;
    const int kb = (lane < 16) ? 0 : 8;
#pragma unroll
    for (int kc = 0; kc < 4; ++kc) {
      const v8bf lo = *(const v8bf*)(qrow + kc * 32 + kb);
      const v8bf hi = *(const v8bf*)(qrow + kc * 32 + kb + 16);
      aq[kc] = cat8(lo, hi);
    }
  }

  v8f o_acc[8];
#pragma unroll
  for (int i = 0; i < 8; ++i)
#pragma unroll
    for (int j = 0; j < 8; ++j) o_acc[i][j] = 0.f;

  float mst[8], lst[8];
#pragma unroll
  for (int r = 0; r < 8; ++r) { mst[r] = -1e30f; lst[r] = 0.f; }

  const float scale = 0.08838834764831845f;   // 1/sqrt(128)
  const int lh = (lane < 16) ? 0 : 8;
  const int nc = lane & 15;
  const int stg_r  = tid & 63;                // staging kv row
  const int stg_d0 = (tid >> 6) * 32;         // staging d segment

  for (int kv0 = 0; kv0 < q0 + 128; kv0 += 64) {
    const __bf16* krow = K + (head_off + kv0 + stg_r) * HD_ + stg_d0;
    const __bf16* vrow = V + (head_off + kv0 + stg_r) * HD_ + stg_d0;

    // Register-stage V (needs LDS-side transpose) while prior tile finishes.
    v8bf rv[4];
#pragma unroll
    for (int j = 0; j < 4; ++j) rv[j] = *(const v8bf*)(vrow + j * 8);
#if !HAVE_ASYNC_LDS
    v8bf rk[4];
#pragma unroll
    for (int j = 0; j < 4; ++j) rk[j] = *(const v8bf*)(krow + j * 8);
#endif

    __syncthreads();                          // previous tile fully consumed

#if HAVE_ASYNC_LDS
    // gfx1250 async copy global->LDS for the K tile (ASYNCcnt tracked).
#pragma unroll
    for (int j = 0; j < 4; ++j)
      __builtin_amdgcn_global_load_async_to_lds_b128(
          (g_vsi4*)(krow + j * 8), (l_vsi4*)&sK[stg_r][stg_d0 + j * 8], 0, 0);
#else
#pragma unroll
    for (int j = 0; j < 4; ++j)
      *(v8bf*)&sK[stg_r][stg_d0 + j * 8] = rk[j];
#endif
#pragma unroll
    for (int j = 0; j < 4; ++j)
#pragma unroll
      for (int e = 0; e < 8; ++e)
        sVt[stg_d0 + j * 8 + e][stg_r] = rv[j][e];

#if HAVE_ASYNC_LDS
    asm volatile("s_wait_asynccnt 0" ::: "memory");
#endif
    __syncthreads();

    // S = Q * K^T  (16 x 64 per wave); batch 4 K-fragments per k-chunk.
    v8f sacc[4];
#pragma unroll
    for (int i = 0; i < 4; ++i)
#pragma unroll
      for (int j = 0; j < 8; ++j) sacc[i][j] = 0.f;
#pragma unroll
    for (int kc = 0; kc < 4; ++kc) {
      v16bf bk[4];
#pragma unroll
      for (int nt = 0; nt < 4; ++nt)
        bk[nt] = load_b_frag(&sK[nt * 16][kc * 32], LDK, lane);
#pragma unroll
      for (int nt = 0; nt < 4; ++nt)
        sacc[nt] = wmma_bf16(aq[kc], bk[nt], sacc[nt]);
    }

    // Online softmax per row (rows confined to 16-lane half-waves).
#pragma unroll
    for (int r = 0; r < 8; ++r) {
      const int qg = q0 + wave * 16 + r + lh;
      float sv[4];
#pragma unroll
      for (int nt = 0; nt < 4; ++nt) {
        const int kg = kv0 + nt * 16 + nc;
        sv[nt] = (kg <= qg) ? (sacc[nt][r] * scale) : -1e30f;
      }
      float rmax = fmaxf(fmaxf(sv[0], sv[1]), fmaxf(sv[2], sv[3]));
#pragma unroll
      for (int off = 1; off < 16; off <<= 1)
        rmax = fmaxf(rmax, __shfl_xor(rmax, off, 32));
      const float mnew  = fmaxf(mst[r], rmax);
      const float alpha = __expf(mst[r] - mnew);
      float psum = 0.f;
#pragma unroll
      for (int nt = 0; nt < 4; ++nt) {
        const float p = __expf(sv[nt] - mnew);
        psum += p;
        sP[wave][r + lh][nt * 16 + nc] = (__bf16)p;
      }
#pragma unroll
      for (int off = 1; off < 16; off <<= 1)
        psum += __shfl_xor(psum, off, 32);
      lst[r] = lst[r] * alpha + psum;
      mst[r] = mnew;
#pragma unroll
      for (int ntd = 0; ntd < 8; ++ntd) o_acc[ntd][r] *= alpha;
    }

    // Wave-local LDS turnaround (LDS pipe is in-order per wave; wait data).
    asm volatile("s_wait_dscnt 0" ::: "memory");

    // O += P * V  (contraction over 64 kv, 2 chunks of 32);
    // batch all 8 V-fragments so ds latency hides behind the WMMA chain.
#pragma unroll
    for (int kc = 0; kc < 2; ++kc) {
      const v16bf ap = load_a_frag(&sP[wave][0][kc * 32], LDV, lane);
      v16bf bv[8];
#pragma unroll
      for (int ntd = 0; ntd < 8; ++ntd)
        bv[ntd] = load_b_frag(&sVt[ntd * 16][kc * 32], LDV, lane);
#pragma unroll
      for (int ntd = 0; ntd < 8; ++ntd)
        o_acc[ntd] = wmma_bf16(ap, bv[ntd], o_acc[ntd]);
    }
  }

  // Normalize and store (B,S,H,hd) == (B*S, DIM) rows for the output GEMM.
#pragma unroll
  for (int r = 0; r < 8; ++r) {
    const int qg = q0 + wave * 16 + r + lh;
    const float rinv = 1.f / lst[r];
#pragma unroll
    for (int ntd = 0; ntd < 8; ++ntd) {
      const int d = ntd * 16 + nc;
      O[((size_t)b * SEQ_ + qg) * DIM_ + h * HD_ + d] =
          (__bf16)(o_acc[ntd][r] * rinv);
    }
  }
}

// ---------------------------------------------------------------------------
// Kernel 3: output projection  out = O * Wo^T  (fp32 result).
// Same pipelined double-buffered structure as kernel 1.
// ---------------------------------------------------------------------------
__global__ __launch_bounds__(256) void out_proj_kernel(
    const __bf16* __restrict__ Ob, const float* __restrict__ Wo,
    float* __restrict__ out)
{
  constexpr int LD = 40;
  __shared__ __bf16 sA[2][128][LD];
  __shared__ __bf16 sB[2][128][LD];

  const int m0   = blockIdx.x * 128;
  const int n0   = blockIdx.y * 128;
  const int tid  = threadIdx.x;
  const int lane = tid & 31;
  const int wave = tid >> 5;

  v8f acc[8];
#pragma unroll
  for (int i = 0; i < 8; ++i)
#pragma unroll
    for (int j = 0; j < 8; ++j) acc[i][j] = 0.f;

  const int rs = tid >> 3;
  const int c4 = (tid & 7) * 4;
  const __bf16* arow = Ob + (size_t)(m0 + rs) * DIM_ + c4;
  const float*  wrow = Wo + (size_t)(n0 + rs) * DIM_ + c4;

  v4bf   ra[4];
  float4 rb[4];
  auto issue = [&](int k0) {
#pragma unroll
    for (int it = 0; it < 4; ++it) {
      ra[it] = *(const v4bf*)(arow + (size_t)it * 32 * DIM_ + k0);
      rb[it] = *(const float4*)(wrow + (size_t)it * 32 * DIM_ + k0);
    }
  };

  issue(0);
  int p = 0;
  for (int kt = 0; kt < DIM_ / 32; ++kt) {
#pragma unroll
    for (int it = 0; it < 4; ++it) {
      const int row = rs + it * 32;
      *(v4bf*)&sA[p][row][c4] = ra[it];
      *(v4bf*)&sB[p][row][c4] = cvt4(rb[it]);
    }
    if (kt + 1 < DIM_ / 32) issue((kt + 1) * 32);
    __syncthreads();
    const v16bf a = load_a_frag(&sA[p][wave * 16][0], LD, lane);
    v16bf bfr[8];
#pragma unroll
    for (int nt = 0; nt < 8; ++nt)
      bfr[nt] = load_b_frag(&sB[p][nt * 16][0], LD, lane);
#pragma unroll
    for (int nt = 0; nt < 8; ++nt)
      acc[nt] = wmma_bf16(a, bfr[nt], acc[nt]);
    p ^= 1;
  }

  const int lh = (lane < 16) ? 0 : 8;
  const int nc = lane & 15;
#pragma unroll
  for (int r = 0; r < 8; ++r) {
    const size_t m = (size_t)(m0 + wave * 16 + r + lh);
#pragma unroll
    for (int nt = 0; nt < 8; ++nt)
      out[m * DIM_ + n0 + nt * 16 + nc] = acc[nt][r];
  }
}

// ---------------------------------------------------------------------------
extern "C" void kernel_launch(void* const* d_in, const int* in_sizes, int n_in,
                              void* d_out, int out_size, void* d_ws, size_t ws_size,
                              hipStream_t stream) {
  (void)in_sizes; (void)n_in; (void)out_size; (void)ws_size;
  const float* x  = (const float*)d_in[0];
  const float* Wq = (const float*)d_in[1];
  const float* Wk = (const float*)d_in[2];
  const float* Wv = (const float*)d_in[3];
  const float* Wo = (const float*)d_in[4];
  float* out = (float*)d_out;

  __bf16* Qb = (__bf16*)d_ws;            // 16 MB each, 64 MB total
  __bf16* Kb = Qb + WSEL_;
  __bf16* Vb = Kb + WSEL_;
  __bf16* Ob = Vb + WSEL_;

  qkv_rope_kernel<<<dim3(32, 16, 3), dim3(256), 0, stream>>>(x, Wq, Wk, Wv,
                                                             Qb, Kb, Vb);
  flash_attn_kernel<<<dim3(16, 16, 2), dim3(256), 0, stream>>>(Qb, Kb, Vb, Ob);
  out_proj_kernel<<<dim3(32, 16, 1), dim3(256), 0, stream>>>(Ob, Wo, out);
}
